// MultiSelf_attention0_38397007626729
// MI455X (gfx1250) — compile-verified, matrix operand order checked by reading, not run on hardware
//
#include <hip/hip_runtime.h>
#include <hip/hip_bf16.h>

// ---------------- problem constants ----------------
#define Bz   4
#define Sz   2048
#define Dz   1024
#define Hz   8
#define DHz  128
#define UTz  1024      // H*DH
#define UNz  128
#define MROWS (Bz*Sz)  // 8192

typedef __attribute__((ext_vector_type(16))) __bf16 v16bf;
typedef __attribute__((ext_vector_type(8)))  float  v8f;

struct alignas(16) u128 { unsigned int w[4]; };
struct alignas(16) u256 { u128 lo, hi; };

// ---------- async global->LDS (gfx1250 ASYNCcnt path), probed ----------
// Probe verdict (round 2 stderr): builtin exists; param 1 is
// 'int __attribute__((vector_size(16))) addrspace(1)*' (printed as __device__*).
#if __has_builtin(__builtin_amdgcn_global_load_async_to_lds_b128)
#define HAVE_ASYNC_LDS 1
typedef int v4i_b128 __attribute__((vector_size(16)));
__device__ __forceinline__ void async_cp16(void* lds, const void* g) {
  __builtin_amdgcn_global_load_async_to_lds_b128(
      (__attribute__((address_space(1))) v4i_b128*)(void*)g,
      (__attribute__((address_space(3))) v4i_b128*)lds,
      /*offset=*/0, /*cpol=*/0);
}
#if __has_builtin(__builtin_amdgcn_s_wait_asynccnt)
#define WAIT_ASYNC() __builtin_amdgcn_s_wait_asynccnt(0)
#else
#define WAIT_ASYNC() asm volatile("s_wait_asynccnt 0x0" ::: "memory")
#endif
#else
#define HAVE_ASYNC_LDS 0
__device__ __forceinline__ void async_cp16(void* lds, const void* g) {
  *reinterpret_cast<u128*>(lds) = *reinterpret_cast<const u128*>(g);
}
#define WAIT_ASYNC() do {} while (0)
#endif

__device__ __forceinline__ unsigned short f32_to_bf16(float f) {
  unsigned int u = __builtin_bit_cast(unsigned int, f);
  unsigned int r = u + 0x7FFFu + ((u >> 16) & 1u);   // round-to-nearest-even
  return (unsigned short)(r >> 16);
}

// Load a 16x32 bf16 WMMA A/B fragment from a row-major bf16 array.
// Lane L: row = L%16; K-halves {0..7,16..23} (L<16) or {8..15,24..31} (L>=16),
// each half is 8 contiguous bf16 = one 16-byte load.
__device__ __forceinline__ v16bf ldfrag(const unsigned short* p, int rowStride) {
  const int lane = threadIdx.x & 31;
  const unsigned short* q = p + (lane & 15) * rowStride + (lane >> 4) * 8;
  u256 t;
  t.lo = *reinterpret_cast<const u128*>(q);
  t.hi = *reinterpret_cast<const u128*>(q + 16);
  return __builtin_bit_cast(v16bf, t);
}

__device__ __forceinline__ v8f wmma_bf16(v16bf a, v16bf b, v8f c) {
  return __builtin_amdgcn_wmma_f32_16x16x32_bf16(
      /*neg_a=*/false, a, /*neg_b=*/false, b,
      /*c_mod=*/(short)0, c, /*reuse_a=*/false, /*reuse_b=*/false);
}

// =====================================================================
// Kernel 1: P = relu(X @ W + b), X:(8192,1024) f32, W:(1024,1024) f32,
// output bf16 (8192,1024). Block tile 128(M) x 64(N), 8 waves of 32x32.
// =====================================================================
__global__ void __launch_bounds__(256)
proj_kernel(const float* __restrict__ X, const float* __restrict__ W,
            const float* __restrict__ bias, unsigned short* __restrict__ P) {
  __shared__ unsigned short Xs[128 * 32];   // [m][k] bf16
  __shared__ unsigned short WsT[64 * 32];   // [n][k] bf16 (transposed)

  const int tid  = threadIdx.x;
  const int lane = tid & 31;
  const int wv   = tid >> 5;
  const int wm   = (wv & 3) * 32;
  const int wn   = (wv >> 2) * 32;
  const int mBase = blockIdx.x * 128;
  const int nBase = blockIdx.y * 64;

  v8f acc[2][2] = {};

  for (int k0 = 0; k0 < Dz; k0 += 32) {
    __syncthreads();
    // stage X tile 128x32 (fp32 -> bf16), coalesced float4 reads
    {
      int r0 = tid >> 3;            // 0..31
      int c4 = (tid & 7) * 4;       // 0..28
      for (int it = 0; it < 4; ++it) {
        int r = r0 + it * 32;
        const float* src = X + (size_t)(mBase + r) * Dz + k0 + c4;
        unsigned int p0 = (unsigned)f32_to_bf16(src[0]) | ((unsigned)f32_to_bf16(src[1]) << 16);
        unsigned int p1 = (unsigned)f32_to_bf16(src[2]) | ((unsigned)f32_to_bf16(src[3]) << 16);
        unsigned int* dst = reinterpret_cast<unsigned int*>(&Xs[r * 32 + c4]);
        dst[0] = p0; dst[1] = p1;
      }
    }
    // stage W tile 32(k)x64(n) transposed into WsT[n][k]
    {
      int kr = tid >> 4;            // 0..15
      int c4 = (tid & 15) * 4;      // 0..60
      for (int it = 0; it < 2; ++it) {
        int kk = kr + it * 16;
        const float* src = W + (size_t)(k0 + kk) * UTz + nBase + c4;
        for (int jj = 0; jj < 4; ++jj)
          WsT[(c4 + jj) * 32 + kk] = f32_to_bf16(src[jj]);
      }
    }
    __syncthreads();

    v16bf a0 = ldfrag(&Xs[(wm +  0) * 32], 32);
    v16bf a1 = ldfrag(&Xs[(wm + 16) * 32], 32);
    v16bf b0 = ldfrag(&WsT[(wn +  0) * 32], 32);
    v16bf b1 = ldfrag(&WsT[(wn + 16) * 32], 32);
    acc[0][0] = wmma_bf16(a0, b0, acc[0][0]);
    acc[0][1] = wmma_bf16(a0, b1, acc[0][1]);
    acc[1][0] = wmma_bf16(a1, b0, acc[1][0]);
    acc[1][1] = wmma_bf16(a1, b1, acc[1][1]);
  }

  // epilogue: bias + relu + bf16 store (C layout: row = v + 8*(lane>>4))
  const int half = lane >> 4;
  const int nlo  = lane & 15;
  for (int j = 0; j < 2; ++j) {
    int n = nBase + wn + j * 16 + nlo;
    float bv = bias[n];
    for (int i = 0; i < 2; ++i) {
      int rbase = mBase + wm + i * 16 + half * 8;
      for (int v = 0; v < 8; ++v) {
        float val = acc[i][j][v] + bv;
        val = val > 0.f ? val : 0.f;
        P[(size_t)(rbase + v) * UTz + n] = f32_to_bf16(val);
      }
    }
  }
}

// =====================================================================
// Kernel 2: flash attention. Block = (b,h) x 128 q-rows; 8 waves x 16 rows.
// Double-buffered LDS staging: K via async global->LDS b128 (ASYNCcnt),
// V prefetched to VGPRs during compute, scattered transposed after.
// =====================================================================
__global__ void __launch_bounds__(256)
flash_kernel(const unsigned short* __restrict__ qp,
             const unsigned short* __restrict__ kp,
             const unsigned short* __restrict__ vp,
             unsigned short* __restrict__ ao) {
  __shared__ unsigned short Ks[2][32 * 128];   // [key][feat], double-buffered
  __shared__ unsigned short VsT[2][128 * 32];  // [feat][key], double-buffered
  __shared__ unsigned short Ps[8][16 * 32];    // per-wave P scratch [row][key]

  const int tid  = threadIdx.x;
  const int lane = tid & 31;
  const int wv   = tid >> 5;
  const int qc   = blockIdx.x & 15;
  const int bh   = blockIdx.x >> 4;
  const int b    = bh >> 3;
  const int h    = bh & 7;
  const int s0   = qc * 128 + wv * 16;
  const size_t rowB = (size_t)b * Sz;

  const unsigned short* kbase = kp + rowB * UTz + h * DHz;
  const unsigned short* vbase = vp + rowB * UTz + h * DHz;

  // staging coordinates for this thread
  const int skey = tid >> 4;          // 0..15
  const int sf8  = (tid & 15) * 8;    // 0..120

  // Q fragments resident in registers: 16 rows x 128 feats = 4 A-frags
  const unsigned short* qbase = qp + (rowB + s0) * UTz + h * DHz;
  v16bf a[4];
  for (int c = 0; c < 4; ++c) a[c] = ldfrag(qbase + c * 32, UTz);

  v8f o[8] = {};
  float m[8], l[8];
  for (int v = 0; v < 8; ++v) { m[v] = -3.0e38f; l[v] = 0.f; }
  const float scale = 0.08838834764831845f;  // 1/sqrt(128)

  // ---- prologue: stage key-block 0 into buffer 0 ----
  for (int it = 0; it < 2; ++it) {
    int kk = skey + it * 16;
    async_cp16(&Ks[0][kk * 128 + sf8], kbase + (size_t)kk * UTz + sf8);
    u128 vv = *reinterpret_cast<const u128*>(vbase + (size_t)kk * UTz + sf8);
    const unsigned short* ve = reinterpret_cast<const unsigned short*>(&vv);
    for (int e = 0; e < 8; ++e) VsT[0][(sf8 + e) * 32 + kk] = ve[e];
  }
  WAIT_ASYNC();

  const int NBLK = Sz / 32;
  for (int blk = 0; blk < NBLK; ++blk) {
    const int cur = blk & 1;
    const int nxt = cur ^ 1;
    __syncthreads();   // publishes buf[cur] staging to all waves

    // kick off next block: K async->LDS, V global->VGPR prefetch
    u128 vpre[2];
    const bool more = (blk + 1 < NBLK);
    if (more) {
      const size_t j1 = (size_t)(blk + 1) * 32;
      for (int it = 0; it < 2; ++it) {
        int kk = skey + it * 16;
        async_cp16(&Ks[nxt][kk * 128 + sf8],
                   kbase + (j1 + kk) * UTz + sf8);
        vpre[it] = *reinterpret_cast<const u128*>(vbase + (j1 + kk) * UTz + sf8);
      }
    }

    // ---- compute on buf[cur] ----
    // S = Q @ K^T : two 16x16 score frags (keys 0..15, 16..31), K-dim 128
    v8f c0 = {}, c1 = {};
    for (int c = 0; c < 4; ++c) {
      v16bf bk0 = ldfrag(&Ks[cur][ 0 * 128 + c * 32], 128);
      v16bf bk1 = ldfrag(&Ks[cur][16 * 128 + c * 32], 128);
      c0 = wmma_bf16(a[c], bk0, c0);
      c1 = wmma_bf16(a[c], bk1, c1);
    }

    // online softmax over this 32-key block (rows live in C layout)
    float alpha[8];
    for (int v = 0; v < 8; ++v) {
      float x0 = c0[v] * scale, x1 = c1[v] * scale;
      float mx = fmaxf(x0, x1);
      mx = fmaxf(mx, __shfl_xor(mx, 1, 32));
      mx = fmaxf(mx, __shfl_xor(mx, 2, 32));
      mx = fmaxf(mx, __shfl_xor(mx, 4, 32));
      mx = fmaxf(mx, __shfl_xor(mx, 8, 32));
      float nm = fmaxf(m[v], mx);
      alpha[v] = __expf(m[v] - nm);
      m[v] = nm;
      float p0 = __expf(x0 - nm);
      float p1 = __expf(x1 - nm);
      c0[v] = p0; c1[v] = p1;
      float rs = p0 + p1;
      rs += __shfl_xor(rs, 1, 32);
      rs += __shfl_xor(rs, 2, 32);
      rs += __shfl_xor(rs, 4, 32);
      rs += __shfl_xor(rs, 8, 32);
      l[v] = l[v] * alpha[v] + rs;
    }
    for (int t = 0; t < 8; ++t)
      for (int v = 0; v < 8; ++v)
        o[t][v] *= alpha[v];

    // C-layout -> A-layout conversion of P through wave-private LDS
    {
      unsigned short* ps = &Ps[wv][0];
      const int half = lane >> 4, nlo = lane & 15;
      for (int v = 0; v < 8; ++v) {
        int r = v + half * 8;
        ps[r * 32 + nlo]      = f32_to_bf16(c0[v]);
        ps[r * 32 + 16 + nlo] = f32_to_bf16(c1[v]);
      }
    }
    asm volatile("s_wait_dscnt 0x0" ::: "memory");  // wave-local LDS RAW fence
    v16bf pA = ldfrag(&Ps[wv][0], 32);

    // O += P @ V : 8 N-tiles of 16, K-dim 32
    for (int t = 0; t < 8; ++t) {
      v16bf vB = ldfrag(&VsT[cur][t * 16 * 32], 32);
      o[t] = wmma_bf16(pA, vB, o[t]);
    }

    // ---- finish staging next block (V scatter + async drain) ----
    if (more) {
      for (int it = 0; it < 2; ++it) {
        int kk = skey + it * 16;
        const unsigned short* ve = reinterpret_cast<const unsigned short*>(&vpre[it]);
        for (int e = 0; e < 8; ++e) VsT[nxt][(sf8 + e) * 32 + kk] = ve[e];
      }
      WAIT_ASYNC();   // next-block K complete before next barrier
    }
  }

  // epilogue: O /= l, store bf16 heads back in (B,S,H*DH) layout
  const int half = lane >> 4, nlo = lane & 15;
  for (int v = 0; v < 8; ++v) {
    float rl = 1.f / l[v];
    int row = s0 + v + half * 8;
    unsigned short* dst = ao + (rowB + row) * UTz + h * DHz;
    for (int t = 0; t < 8; ++t)
      dst[t * 16 + nlo] = f32_to_bf16(o[t][v] * rl);
  }
}

// =====================================================================
// Kernel 3: Out = AO @ W0 + b0.  AO:(8192,1024) bf16, W0:(1024,128) f32,
// Out:(8192,128) f32.  Same tiling as proj_kernel.
// =====================================================================
__global__ void __launch_bounds__(256)
out_kernel(const unsigned short* __restrict__ AO, const float* __restrict__ W0,
           const float* __restrict__ bias, float* __restrict__ Out) {
  __shared__ unsigned short Xs[128 * 32];
  __shared__ unsigned short WsT[64 * 32];

  const int tid  = threadIdx.x;
  const int lane = tid & 31;
  const int wv   = tid >> 5;
  const int wm   = (wv & 3) * 32;
  const int wn   = (wv >> 2) * 32;
  const int mBase = blockIdx.x * 128;
  const int nBase = blockIdx.y * 64;

  v8f acc[2][2] = {};

  for (int k0 = 0; k0 < UTz; k0 += 32) {
    __syncthreads();
    // stage AO tile 128x32 bf16 (already bf16, b128 copies)
    {
      int r0 = tid >> 2;            // 0..63
      int f8 = (tid & 3) * 8;       // 0..24
      for (int it = 0; it < 2; ++it) {
        int r = r0 + it * 64;
        *reinterpret_cast<u128*>(&Xs[r * 32 + f8]) =
            *reinterpret_cast<const u128*>(AO + (size_t)(mBase + r) * UTz + k0 + f8);
      }
    }
    // stage W0 tile 32(k)x64(n) transposed (fp32 -> bf16)
    {
      int kr = tid >> 4;
      int c4 = (tid & 15) * 4;
      for (int it = 0; it < 2; ++it) {
        int kk = kr + it * 16;
        const float* src = W0 + (size_t)(k0 + kk) * UNz + nBase + c4;
        for (int jj = 0; jj < 4; ++jj)
          WsT[(c4 + jj) * 32 + kk] = f32_to_bf16(src[jj]);
      }
    }
    __syncthreads();

    v16bf a0 = ldfrag(&Xs[(wm +  0) * 32], 32);
    v16bf a1 = ldfrag(&Xs[(wm + 16) * 32], 32);
    v16bf b0 = ldfrag(&WsT[(wn +  0) * 32], 32);
    v16bf b1 = ldfrag(&WsT[(wn + 16) * 32], 32);
    acc[0][0] = wmma_bf16(a0, b0, acc[0][0]);
    acc[0][1] = wmma_bf16(a0, b1, acc[0][1]);
    acc[1][0] = wmma_bf16(a1, b0, acc[1][0]);
    acc[1][1] = wmma_bf16(a1, b1, acc[1][1]);
  }

  const int half = lane >> 4;
  const int nlo  = lane & 15;
  for (int j = 0; j < 2; ++j) {
    int n = nBase + wn + j * 16 + nlo;
    float bv = bias[n];
    for (int i = 0; i < 2; ++i) {
      int rbase = mBase + wm + i * 16 + half * 8;
      for (int v = 0; v < 8; ++v)
        Out[(size_t)(rbase + v) * UNz + n] = acc[i][j][v] + bv;
    }
  }
}

// =====================================================================
extern "C" void kernel_launch(void* const* d_in, const int* in_sizes, int n_in,
                              void* d_out, int out_size, void* d_ws, size_t ws_size,
                              hipStream_t stream) {
  const float* q   = (const float*)d_in[0];
  const float* k   = (const float*)d_in[1];
  const float* v   = (const float*)d_in[2];
  const float* w_q = (const float*)d_in[3];
  const float* b_q = (const float*)d_in[4];
  const float* w_k = (const float*)d_in[5];
  const float* b_k = (const float*)d_in[6];
  const float* w_v = (const float*)d_in[7];
  const float* b_v = (const float*)d_in[8];
  const float* w_0 = (const float*)d_in[9];
  const float* b_0 = (const float*)d_in[10];

  // workspace: qp | kp | vp | ao, each 8192x1024 bf16 = 16 MB (64 MB total)
  unsigned short* qp = (unsigned short*)d_ws;
  unsigned short* kp = qp + (size_t)MROWS * UTz;
  unsigned short* vp = kp + (size_t)MROWS * UTz;
  unsigned short* ao = vp + (size_t)MROWS * UTz;

  dim3 blk(256);
  dim3 gProj(MROWS / 128, UTz / 64);            // 64 x 16
  proj_kernel<<<gProj, blk, 0, stream>>>(q, w_q, b_q, qp);
  proj_kernel<<<gProj, blk, 0, stream>>>(k, w_k, b_k, kp);
  proj_kernel<<<gProj, blk, 0, stream>>>(v, w_v, b_v, vp);

  flash_kernel<<<Bz * Hz * (Sz / 128), blk, 0, stream>>>(qp, kp, vp, ao);

  dim3 gOut(MROWS / 128, UNz / 64);             // 64 x 2
  out_kernel<<<gOut, blk, 0, stream>>>(ao, w_0, b_0, (float*)d_out);
}
